// MambaEncoder_77086073028640
// MI455X (gfx1250) — matrix operand order repaired and checked by reading, not compiled, not run on hardware
//
#include <hip/hip_runtime.h>

// ---------- Problem constants ----------
#define BATCH   2
#define SEQ     2048
#define NROWS   (BATCH*SEQ)       // 4096
#define DMODEL  768
#define DINNER  1536
#define DSTATE  16
#define DCONV   4
#define DTRANK  48
#define DTPAD   64                // DTRANK zero-padded to WMMA-friendly K
#define XPROJC  (DTRANK + 2*DSTATE) // 80

// ---------- WMMA types ----------
typedef __attribute__((ext_vector_type(16))) __bf16          v16bf;
typedef __attribute__((ext_vector_type(8)))  float           v8f;
typedef __attribute__((ext_vector_type(8)))  unsigned short  u16x8;
typedef __attribute__((ext_vector_type(16))) unsigned short  u16x16;

__device__ __forceinline__ unsigned short f2bf(float f) {
    unsigned u = __builtin_bit_cast(unsigned, f);
    u += 0x7fffu + ((u >> 16) & 1u);          // round-to-nearest-even
    return (unsigned short)(u >> 16);
}

__device__ __forceinline__ v16bf mk_frag(u16x8 lo, u16x8 hi) {
    u16x16 t;
#pragma unroll
    for (int i = 0; i < 8; ++i) { t[i] = lo[i]; t[i + 8] = hi[i]; }
    return __builtin_bit_cast(v16bf, t);
}

// ---------- Weight transpose + fp32->bf16 (dst [N, Kpad], src [K, N]) ----------
__global__ void k_tconv(const float* __restrict__ src, unsigned short* __restrict__ dst,
                        int K, int N, int Kpad) {
    int idx = blockIdx.x * blockDim.x + threadIdx.x;
    if (idx >= N * Kpad) return;
    int n = idx / Kpad, k = idx % Kpad;
    dst[idx] = (k < K) ? f2bf(src[(size_t)k * N + n]) : (unsigned short)0;
}

// ---------- LayerNorm -> bf16 row [NROWS, DMODEL] ----------
__global__ __launch_bounds__(256) void k_layernorm(const float* __restrict__ x,
                                                   const float* __restrict__ g,
                                                   const float* __restrict__ b,
                                                   unsigned short* __restrict__ xn) {
    __shared__ float s1[256], s2[256];
    const int row = blockIdx.x, tid = threadIdx.x;
    const float* xp = x + (size_t)row * DMODEL;
    float sum = 0.f, sq = 0.f;
    for (int i = tid; i < DMODEL; i += 256) { float v = xp[i]; sum += v; sq += v * v; }
    s1[tid] = sum; s2[tid] = sq; __syncthreads();
    for (int off = 128; off > 0; off >>= 1) {
        if (tid < off) { s1[tid] += s1[tid + off]; s2[tid] += s2[tid + off]; }
        __syncthreads();
    }
    const float mu  = s1[0] * (1.f / DMODEL);
    const float var = s2[0] * (1.f / DMODEL) - mu * mu;
    const float inv = rsqrtf(var + 1e-5f);
    for (int i = tid; i < DMODEL; i += 256)
        xn[(size_t)row * DMODEL + i] = f2bf((xp[i] - mu) * inv * g[i] + b[i]);
}

// ---------- WMMA GEMM helpers ----------
__device__ __forceinline__ void load_frags(const unsigned short* const arow[4],
                                           const unsigned short* const brow[2],
                                           int kb, v16bf af[4], v16bf bfr[2]) {
#pragma unroll
    for (int im = 0; im < 4; ++im) {
        const unsigned short* p = arow[im] + kb;
        af[im] = mk_frag(*(const u16x8*)p, *(const u16x8*)(p + 16));
    }
#pragma unroll
    for (int jn = 0; jn < 2; ++jn) {
        const unsigned short* p = brow[jn] + kb;
        bfr[jn] = mk_frag(*(const u16x8*)p, *(const u16x8*)(p + 16));
    }
}

__device__ __forceinline__ void do_wmma(v8f acc[4][2], const v16bf af[4], const v16bf bfr[2]) {
#pragma unroll
    for (int im = 0; im < 4; ++im)
#pragma unroll
        for (int jn = 0; jn < 2; ++jn)
            acc[im][jn] = __builtin_amdgcn_wmma_f32_16x16x32_bf16(
                false, af[im], false, bfr[jn], (short)0, acc[im][jn], false, false);
}

// ---------- WMMA bf16 GEMM: C[M,N] = A[M,K] @ Bt[N,K]^T  (fp32 accumulate) ----------
// Block tile 128x128: 8 waves as 2 (M) x 4 (N); wave tile 64x32 = 4x2 WMMA tiles.
// K-loop is double-buffered in 2x32 steps (all K here are multiples of 64; a
// 32-wide tail keeps it generic). epi: 0=none, 1=+resid[m,n], 2=softplus(acc+bias[n])
#define TM 128
#define TN 128
__global__ __launch_bounds__(256) void k_gemm(
    const unsigned short* __restrict__ A,
    const unsigned short* __restrict__ Bt,
    float* __restrict__ C, int M, int N, int K,
    int epi, const float* __restrict__ bias, const float* __restrict__ resid)
{
    const int tid  = threadIdx.x;
    const int lane = tid & 31, wave = tid >> 5;
    const int wm = wave >> 2, wn = wave & 3;                 // 2 (M) x 4 (N) wave grid
    const int m0 = blockIdx.y * TM + wm * 64;
    const int n0 = blockIdx.x * TN + wn * 32;
    const int l16 = lane & 15, lh = lane >> 4;               // lane-half select
    const int ka = lh * 8;                                   // A: K-half offset (elems)
    const int kb16 = lh * 8;                                 // B: K-half offset (elems)

    v8f acc[4][2];
#pragma unroll
    for (int i = 0; i < 4; ++i)
#pragma unroll
        for (int j = 0; j < 2; ++j)
#pragma unroll
            for (int e = 0; e < 8; ++e) acc[i][j][e] = 0.f;

    // Hoisted row base pointers. B rows for dead (>=N) output columns are
    // clamped to row N-1: loads stay in-bounds, results are never stored.
    const unsigned short* arow[4];
#pragma unroll
    for (int im = 0; im < 4; ++im)
        arow[im] = A + (size_t)(m0 + im * 16 + l16) * K + ka;
    const unsigned short* brow[2];
#pragma unroll
    for (int jn = 0; jn < 2; ++jn) {
        int n = n0 + jn * 16 + l16;
        if (n > N - 1) n = N - 1;
        brow[jn] = Bt + (size_t)n * K + kb16;
    }

    // Software-pipelined K loop: ping-pong fragment buffers.
    v16bf af0[4], bf0[2], af1[4], bf1[2];
    load_frags(arow, brow, 0, af0, bf0);
    int kb = 0;
    for (; kb + 64 <= K; kb += 64) {
        load_frags(arow, brow, kb + 32, af1, bf1);
        __builtin_prefetch(arow[0] + kb + 64, 0, 1);          // speculative: OOB is dropped
        __builtin_prefetch(arow[2] + kb + 64, 0, 1);
        __builtin_prefetch(brow[0] + kb + 64, 0, 1);
        do_wmma(acc, af0, bf0);
        if (kb + 64 < K) load_frags(arow, brow, kb + 64, af0, bf0);
        do_wmma(acc, af1, bf1);
    }
    if (kb < K)                                               // K % 64 == 32 tail
        do_wmma(acc, af0, bf0);

    // C/D layout: VGPR r -> M = r + (lane/16)*8 ; N = lane%16
#pragma unroll
    for (int im = 0; im < 4; ++im) {
#pragma unroll
        for (int jn = 0; jn < 2; ++jn) {
            const int ncol = n0 + jn * 16 + l16;
            if (ncol >= N) continue;
#pragma unroll
            for (int r = 0; r < 8; ++r) {
                const int mrow = m0 + im * 16 + lh * 8 + r;
                float v = acc[im][jn][r];
                const size_t idx = (size_t)mrow * N + ncol;
                if (epi == 1) v += resid[idx];
                else if (epi == 2) {
                    v += bias[ncol];
                    v = (v > 20.f) ? v : log1pf(__expf(v));   // softplus
                }
                C[idx] = v;
            }
        }
    }
}

// ---------- Depthwise causal conv (DCONV=4) + SiLU; reads xs-half of xr ----------
__global__ void k_conv_silu(const float* __restrict__ xr, const float* __restrict__ cw,
                            const float* __restrict__ cb, float* __restrict__ xs,
                            unsigned short* __restrict__ xsb) {
    const int idx = blockIdx.x * blockDim.x + threadIdx.x;
    if (idx >= NROWS * DINNER) return;
    const int d = idx % DINNER;
    const int row = idx / DINNER;
    const int l = row % SEQ;
    float acc = cb[d];
#pragma unroll
    for (int k = 0; k < DCONV; ++k) {
        const int ll = l - (DCONV - 1) + k;
        if (ll >= 0) acc += cw[d * DCONV + k] * xr[(size_t)(row - (DCONV - 1) + k) * (2 * DINNER) + d];
    }
    const float s = acc * (1.f / (1.f + __expf(-acc)));       // silu
    xs[idx] = s;
    xsb[idx] = f2bf(s);
}

// ---------- dt slice of dbl -> zero-padded bf16 [NROWS, DTPAD] ----------
__global__ void k_dtpad(const float* __restrict__ dbl, unsigned short* __restrict__ dtb) {
    const int idx = blockIdx.x * blockDim.x + threadIdx.x;
    if (idx >= NROWS * DTPAD) return;
    const int r = idx >> 6, j = idx & 63;
    dtb[idx] = (j < DTRANK) ? f2bf(dbl[(size_t)r * XPROJC + j]) : (unsigned short)0;
}

// ---------- Selective scan: one lane per (b,d); 16-state in registers ----------
__global__ __launch_bounds__(256) void k_scan(
    const float* __restrict__ delta, const float* __restrict__ xs,
    const float* __restrict__ dbl,   const float* __restrict__ xr,
    const float* __restrict__ A_log, const float* __restrict__ Dskip,
    unsigned short* __restrict__ ybf)
{
    const int gid = blockIdx.x * blockDim.x + threadIdx.x;
    if (gid >= BATCH * DINNER) return;
    const int b = gid / DINNER, d = gid % DINNER;
    float A[DSTATE], h[DSTATE];
#pragma unroll
    for (int n = 0; n < DSTATE; ++n) { A[n] = -__expf(A_log[d * DSTATE + n]); h[n] = 0.f; }
    const float dsk = Dskip[d];
    for (int l = 0; l < SEQ; ++l) {
        const size_t row = (size_t)b * SEQ + l;
        const float dv = delta[row * DINNER + d];
        const float xv = xs[row * DINNER + d];
        const float* bc = dbl + row * XPROJC;
        float y = 0.f;
#pragma unroll
        for (int n = 0; n < DSTATE; ++n) {
            const float dA = __expf(dv * A[n]);
            h[n] = dA * h[n] + dv * bc[DTRANK + n] * xv;      // dBu
            y += h[n] * bc[DTRANK + DSTATE + n];               // * C
        }
        const float res = xr[row * (2 * DINNER) + DINNER + d];
        y = (y + xv * dsk) * (res * (1.f / (1.f + __expf(-res))));
        ybf[row * DINNER + d] = f2bf(y);
    }
}

// ---------- Host orchestration ----------
extern "C" void kernel_launch(void* const* d_in, const int* in_sizes, int n_in,
                              void* d_out, int out_size, void* d_ws, size_t ws_size,
                              hipStream_t stream) {
    const float* x       = (const float*)d_in[0];
    const float* gamma   = (const float*)d_in[1];
    const float* beta    = (const float*)d_in[2];
    const float* W_in    = (const float*)d_in[3];
    const float* conv_w  = (const float*)d_in[4];
    const float* conv_b  = (const float*)d_in[5];
    const float* W_xproj = (const float*)d_in[6];
    const float* W_dt    = (const float*)d_in[7];
    const float* b_dt    = (const float*)d_in[8];
    const float* A_log   = (const float*)d_in[9];
    const float* D_skip  = (const float*)d_in[10];
    const float* W_out   = (const float*)d_in[11];
    float* out = (float*)d_out;

    char* w = (char*)d_ws;
    size_t off = 0;
    auto alloc = [&](size_t bytes) -> void* {
        void* p = w + off;
        off = (off + bytes + 255) & ~(size_t)255;
        return p;
    };
    unsigned short* xn_b  = (unsigned short*)alloc((size_t)NROWS * DMODEL * 2);
    float*          xr    = (float*)         alloc((size_t)NROWS * 2 * DINNER * 4);
    float*          xs_f  = (float*)         alloc((size_t)NROWS * DINNER * 4);
    unsigned short* xs_b  = (unsigned short*)alloc((size_t)NROWS * DINNER * 2);
    float*          dbl   = (float*)         alloc((size_t)NROWS * XPROJC * 4);
    unsigned short* dt_b  = (unsigned short*)alloc((size_t)NROWS * DTPAD * 2);
    float*          delta = (float*)         alloc((size_t)NROWS * DINNER * 4);
    unsigned short* y_b   = (unsigned short*)alloc((size_t)NROWS * DINNER * 2);
    unsigned short* wtin  = (unsigned short*)alloc((size_t)2 * DINNER * DMODEL * 2);
    unsigned short* wtxp  = (unsigned short*)alloc((size_t)XPROJC * DINNER * 2);
    unsigned short* wtdt  = (unsigned short*)alloc((size_t)DINNER * DTPAD * 2);
    unsigned short* wtout = (unsigned short*)alloc((size_t)DMODEL * DINNER * 2);

    for (int i = 0; i < 2; ++i) {
        const float* g_i   = gamma   + (size_t)i * DMODEL;
        const float* b_i   = beta    + (size_t)i * DMODEL;
        const float* Win_i = W_in    + (size_t)i * DMODEL * 2 * DINNER;
        const float* cw_i  = conv_w  + (size_t)i * DINNER * DCONV;
        const float* cb_i  = conv_b  + (size_t)i * DINNER;
        const float* Wxp_i = W_xproj + (size_t)i * DINNER * XPROJC;
        const float* Wdt_i = W_dt    + (size_t)i * DTRANK * DINNER;
        const float* bdt_i = b_dt    + (size_t)i * DINNER;
        const float* Al_i  = A_log   + (size_t)i * DINNER * DSTATE;
        const float* Ds_i  = D_skip  + (size_t)i * DINNER;
        const float* Wo_i  = W_out   + (size_t)i * DINNER * DMODEL;

        const float* xin  = (i == 0) ? x : out;
        float*       xout = out;

        // Weight transpose/convert to bf16 [N, Kpad]
        {
            int n;
            n = 2 * DINNER * DMODEL;
            k_tconv<<<(n + 255) / 256, 256, 0, stream>>>(Win_i, wtin, DMODEL, 2 * DINNER, DMODEL);
            n = XPROJC * DINNER;
            k_tconv<<<(n + 255) / 256, 256, 0, stream>>>(Wxp_i, wtxp, DINNER, XPROJC, DINNER);
            n = DINNER * DTPAD;
            k_tconv<<<(n + 255) / 256, 256, 0, stream>>>(Wdt_i, wtdt, DTRANK, DINNER, DTPAD);
            n = DMODEL * DINNER;
            k_tconv<<<(n + 255) / 256, 256, 0, stream>>>(Wo_i, wtout, DINNER, DMODEL, DINNER);
        }

        // LayerNorm -> bf16
        k_layernorm<<<NROWS, 256, 0, stream>>>(xin, g_i, b_i, xn_b);

        // xr = xn @ W_in   [4096 x 3072], K=768
        {
            dim3 grid((2 * DINNER + TN - 1) / TN, (NROWS + TM - 1) / TM);
            k_gemm<<<grid, 256, 0, stream>>>(xn_b, wtin, xr, NROWS, 2 * DINNER, DMODEL,
                                             0, nullptr, nullptr);
        }

        // depthwise causal conv + silu
        {
            int n = NROWS * DINNER;
            k_conv_silu<<<(n + 255) / 256, 256, 0, stream>>>(xr, cw_i, cb_i, xs_f, xs_b);
        }

        // dbl = xs @ W_xproj   [4096 x 80], K=1536
        {
            dim3 grid((XPROJC + TN - 1) / TN, (NROWS + TM - 1) / TM);
            k_gemm<<<grid, 256, 0, stream>>>(xs_b, wtxp, dbl, NROWS, XPROJC, DINNER,
                                             0, nullptr, nullptr);
        }

        // dt -> padded bf16
        {
            int n = NROWS * DTPAD;
            k_dtpad<<<(n + 255) / 256, 256, 0, stream>>>(dbl, dt_b);
        }

        // delta = softplus(dt @ W_dt + b_dt)   [4096 x 1536], K=64
        {
            dim3 grid((DINNER + TN - 1) / TN, (NROWS + TM - 1) / TM);
            k_gemm<<<grid, 256, 0, stream>>>(dt_b, wtdt, delta, NROWS, DINNER, DTPAD,
                                             2, bdt_i, nullptr);
        }

        // selective scan + D-skip + silu(res) gate -> y bf16
        {
            int n = BATCH * DINNER;
            k_scan<<<(n + 255) / 256, 256, 0, stream>>>(delta, xs_f, dbl, xr, Al_i, Ds_i, y_b);
        }

        // x_next = x + y @ W_out   [4096 x 768], K=1536
        {
            dim3 grid((DMODEL + TN - 1) / TN, (NROWS + TM - 1) / TM);
            k_gemm<<<grid, 256, 0, stream>>>(y_b, wtout, xout, NROWS, DMODEL, DINNER,
                                             1, nullptr, xin);
        }
    }
    (void)in_sizes; (void)n_in; (void)out_size; (void)ws_size;
}
